// VSAMoEBlock_84181359001993
// MI455X (gfx1250) — compile-verified, matrix operand order checked by reading, not run on hardware
//
#include <hip/hip_runtime.h>
#include <hip/hip_bf16.h>

typedef __attribute__((ext_vector_type(16))) __bf16 v16bf;
typedef __attribute__((ext_vector_type(8)))  __bf16 v8bf;
typedef __attribute__((ext_vector_type(8)))  float  v8f;

static constexpr int NTOK = 8192;   // B*T
static constexpr int DDIM = 2048;
static constexpr int FDIM = 5632;
static constexpr int NEXP = 8;

#define TM 64
#define TN 128
#define TK 32
#define TKP 40   // padded K-stride (elements): 80B rows -> conflict-free b128 frag reads

__device__ __forceinline__ __bf16 f2bf(float f) {
  unsigned u = __builtin_bit_cast(unsigned, f);
  u += 0x7FFFu + ((u >> 16) & 1u);          // round-to-nearest-even
  unsigned short s = (unsigned short)(u >> 16);
  return __builtin_bit_cast(__bf16, s);
}

__device__ __forceinline__ unsigned pack2bf(float a, float b) {
  unsigned ua = __builtin_bit_cast(unsigned, a);
  unsigned ub = __builtin_bit_cast(unsigned, b);
  ua += 0x7FFFu + ((ua >> 16) & 1u);
  ub += 0x7FFFu + ((ub >> 16) & 1u);
  return (ua >> 16) | (ub & 0xFFFF0000u);
}

// CDNA5 async memory->LDS copy, 16B per lane, tracked on ASYNCcnt (ISA 15.18.3 op 98).
__device__ __forceinline__ void async_copy_b128(const __bf16* gsrc, __bf16* lds_dst) {
  unsigned laddr = (unsigned)(uintptr_t)lds_dst;   // low 32 bits of generic ptr = LDS byte addr
  asm volatile("global_load_async_to_lds_b128 %0, %1, off"
               :: "v"(laddr), "v"(gsrc) : "memory");
}

// Async loads complete in order: waiting "<= N outstanding" means all but the
// most recent N have landed in LDS.
__device__ __forceinline__ void wait_async_le5() { asm volatile("s_wait_asynccnt 0x5" ::: "memory"); }
__device__ __forceinline__ void wait_async_le3() { asm volatile("s_wait_asynccnt 0x3" ::: "memory"); }
__device__ __forceinline__ void wait_async_0()   { asm volatile("s_wait_asynccnt 0x0" ::: "memory"); }

// v16bf fragment: elements 0..7 = K kb..kb+7, 8..15 = K kb+16..kb+23
// (ISA 7.12.2 16-bit operand layout), from contiguous-K LDS storage.
__device__ __forceinline__ v16bf load_frag(const __bf16* base) {
  v8bf lo = *(const v8bf*)(base);
  v8bf hi = *(const v8bf*)(base + 16);
  v16bf r;
#pragma unroll
  for (int j = 0; j < 8; ++j) { r[j] = lo[j]; r[j + 8] = hi[j]; }
  return r;
}

// ---------------- prep kernels ----------------

__global__ void cast_x_k(const float* __restrict__ x, __bf16* __restrict__ xb, size_t n4) {
  size_t i = (size_t)blockIdx.x * blockDim.x + threadIdx.x;
  if (i < n4) {
    float4 v = ((const float4*)x)[i];
    uint2 p; p.x = pack2bf(v.x, v.y); p.y = pack2bf(v.z, v.w);
    *(uint2*)(xb + i * 4) = p;
  }
}

__global__ void zero_f4_k(float* __restrict__ p, size_t n4) {
  size_t i = (size_t)blockIdx.x * blockDim.x + threadIdx.x;
  if (i < n4) ((float4*)p)[i] = make_float4(0.f, 0.f, 0.f, 0.f);
}

__global__ void zero_i_k(int* __restrict__ p, int n) {
  int i = blockIdx.x * blockDim.x + threadIdx.x;
  if (i < n) p[i] = 0;
}

// Transpose + transcode: src [R, C] fp32 -> dst [C, R] bf16, one expert per blockIdx.z.
__global__ __launch_bounds__(256) void transcode_T_k(const float* __restrict__ src,
                                                     __bf16* __restrict__ dst,
                                                     int R, int C) {
  __shared__ float tile[32][33];
  const size_t eoff = (size_t)blockIdx.z * R * C;
  const int r0 = blockIdx.y * 32, c0 = blockIdx.x * 32;
  const int tid = threadIdx.x;
  const int tr = tid >> 3;             // 0..31
  const int tc4 = (tid & 7) << 2;      // 0,4,..,28
  float4 v = *(const float4*)(src + eoff + (size_t)(r0 + tr) * C + c0 + tc4);
  tile[tr][tc4 + 0] = v.x; tile[tr][tc4 + 1] = v.y;
  tile[tr][tc4 + 2] = v.z; tile[tr][tc4 + 3] = v.w;
  __syncthreads();
  // dst row = c0+tr, cols r0+tc4..+3 (K-contiguous output)
  uint2 p;
  p.x = pack2bf(tile[tc4 + 0][tr], tile[tc4 + 1][tr]);
  p.y = pack2bf(tile[tc4 + 2][tr], tile[tc4 + 3][tr]);
  *(uint2*)(dst + eoff + (size_t)(c0 + tr) * R + r0 + tc4) = p;
}

__global__ void routing_k(const float* __restrict__ w, const int* __restrict__ idx,
                          int* __restrict__ counts, int* __restrict__ tok_ids,
                          float* __restrict__ wkbuf) {
  int n = blockIdx.x * blockDim.x + threadIdx.x;
  if (n >= NTOK) return;
  int e0 = idx[n * 2 + 0], e1 = idx[n * 2 + 1];
  float w0 = w[n * 2 + 0], w1 = w[n * 2 + 1];
  if (e0 == e1) w0 += w1;                 // merge duplicate-expert slots
  int p = atomicAdd(&counts[e0], 1);
  tok_ids[e0 * NTOK + p] = n; wkbuf[e0 * NTOK + p] = w0;
  if (e1 != e0) {
    p = atomicAdd(&counts[e1], 1);
    tok_ids[e1 * NTOK + p] = n; wkbuf[e1 * NTOK + p] = w1;
  }
}

__global__ void scan_k(const int* __restrict__ counts, int* __restrict__ offsets) {
  if (threadIdx.x == 0) {
    int s = 0;
    for (int e = 0; e < NEXP; ++e) { offsets[e] = s; s += counts[e]; }
    offsets[NEXP] = s;
  }
}

// ---------------- GEMM1: h = silu(x@gate) * (x@up) ----------------

__global__ __launch_bounds__(256) void gemm1_kernel(
    const __bf16* __restrict__ xb,       // [NTOK, D] bf16
    const __bf16* __restrict__ gate_bt,  // [E, F, D] bf16 (K-contiguous)
    const __bf16* __restrict__ up_bt,    // [E, F, D] bf16
    const int* __restrict__ counts,
    const int* __restrict__ offsets,
    const int* __restrict__ tok_ids,     // [E, NTOK]
    __bf16* __restrict__ hbuf)           // [totalEntries, F] bf16
{
  const int e = blockIdx.z;
  const int cnt = counts[e];
  const int m0 = blockIdx.y * TM;
  if (m0 >= cnt) return;
  const int n0 = blockIdx.x * TN;
  const int hbase = offsets[e];

  __shared__ __bf16 As[2][TM][TKP];     // ping-pong double buffers
  __shared__ __bf16 Bg[2][TN][TKP];
  __shared__ __bf16 Bu[2][TN][TKP];
  __shared__ int rowTok[TM];

  const int tid = threadIdx.x;
  if (tid < TM) {
    int m = m0 + tid;
    rowTok[tid] = tok_ids[e * NTOK + ((m < cnt) ? m : (cnt - 1))];
  }
  __syncthreads();

  const int wave = tid >> 5, lane = tid & 31;
  const int mw = wave >> 1, nw = wave & 1;
  const int arow = (mw << 4) + (lane & 15);
  const int kb = (lane & 16) ? 8 : 0;

  v8f zz = {};
  v8f accg[4], accu[4];
#pragma unroll
  for (int s = 0; s < 4; ++s) { accg[s] = zz; accu[s] = zz; }

  // copy roles: A tile 64x32 (1 x b128/thread); B tiles 128x32 (2 x b128/thread each)
  const int lr = tid >> 2, lc = (tid & 3) << 3;
  const int bc = tid >> 2, bk = (tid & 3) << 3;
  const __bf16* aptr  = xb + (size_t)rowTok[lr] * DDIM + lc;
  const __bf16* gcol0 = gate_bt + ((size_t)e * FDIM + n0 + bc) * DDIM + bk;
  const __bf16* gcol1 = gcol0 + (size_t)64 * DDIM;
  const __bf16* ucol0 = up_bt + ((size_t)e * FDIM + n0 + bc) * DDIM + bk;
  const __bf16* ucol1 = ucol0 + (size_t)64 * DDIM;

  auto issue = [&](int k0, int p) {
    async_copy_b128(aptr + k0,  &As[p][lr][lc]);
    async_copy_b128(gcol0 + k0, &Bg[p][bc][bk]);
    async_copy_b128(gcol1 + k0, &Bg[p][bc + 64][bk]);
    async_copy_b128(ucol0 + k0, &Bu[p][bc][bk]);
    async_copy_b128(ucol1 + k0, &Bu[p][bc + 64][bk]);
  };

  auto compute = [&](int p) {
    v16bf af = load_frag(&As[p][arow][kb]);
    v16bf bg[4], bu[4];
#pragma unroll
    for (int s = 0; s < 4; ++s) {
      int bcc = (nw << 6) + (s << 4) + (lane & 15);
      bg[s] = load_frag(&Bg[p][bcc][kb]);
      bu[s] = load_frag(&Bu[p][bcc][kb]);
    }
#pragma unroll
    for (int s = 0; s < 4; ++s) {
      accg[s] = __builtin_amdgcn_wmma_f32_16x16x32_bf16(false, af, false, bg[s], (short)0, accg[s], false, false);
      accu[s] = __builtin_amdgcn_wmma_f32_16x16x32_bf16(false, af, false, bu[s], (short)0, accu[s], false, false);
    }
  };

  // branch-free unroll-by-2 pipeline; NK = 64 tiles (even), last pair peeled
  constexpr int NK = DDIM / TK;
  issue(0, 0);
  for (int t = 0; t < NK - 2; t += 2) {
    issue((t + 1) * TK, 1);
    wait_async_le5();
    __syncthreads();
    compute(0);
    __syncthreads();
    issue((t + 2) * TK, 0);
    wait_async_le5();
    __syncthreads();
    compute(1);
    __syncthreads();
  }
  issue((NK - 1) * TK, 1);
  wait_async_le5();
  __syncthreads();
  compute(0);
  __syncthreads();
  wait_async_0();
  __syncthreads();
  compute(1);

#pragma unroll
  for (int s = 0; s < 4; ++s) {
    int col = n0 + (nw << 6) + (s << 4) + (lane & 15);
#pragma unroll
    for (int j = 0; j < 8; ++j) {
      int row = (mw << 4) + ((lane & 16) ? (j + 8) : j);
      int m = m0 + row;
      if (m < cnt) {
        float g = accg[s][j];
        float u = accu[s][j];
        float hv = g * __builtin_amdgcn_rcpf(1.0f + __expf(-g)) * u;  // silu(g)*u, HW rcp
        hbuf[(size_t)(hbase + m) * FDIM + col] = f2bf(hv);
      }
    }
  }
}

// ---------------- GEMM2: out[tok] += wk * (h @ down_w[e]) ----------------

__global__ __launch_bounds__(256) void gemm2_kernel(
    const __bf16* __restrict__ hbuf,     // [totalEntries, F]
    const __bf16* __restrict__ down_bt,  // [E, D, F] bf16 (K-contiguous)
    const int* __restrict__ counts,
    const int* __restrict__ offsets,
    const int* __restrict__ tok_ids,
    const float* __restrict__ wkbuf,
    float* __restrict__ out)             // [NTOK, D] fp32
{
  const int e = blockIdx.z;
  const int cnt = counts[e];
  const int m0 = blockIdx.y * TM;
  if (m0 >= cnt) return;
  const int n0 = blockIdx.x * TN;
  const int hbase = offsets[e];

  __shared__ __bf16 As[2][TM][TKP];
  __shared__ __bf16 Bd[2][TN][TKP];
  __shared__ int rowTok[TM];
  __shared__ float rowWk[TM];

  const int tid = threadIdx.x;
  if (tid < TM) {
    int m = m0 + tid;
    int mm = (m < cnt) ? m : (cnt - 1);
    rowTok[tid] = tok_ids[e * NTOK + mm];
    rowWk[tid] = wkbuf[e * NTOK + mm];
  }
  __syncthreads();

  const int wave = tid >> 5, lane = tid & 31;
  const int mw = wave >> 1, nw = wave & 1;
  const int arow = (mw << 4) + (lane & 15);
  const int kb = (lane & 16) ? 8 : 0;

  v8f zz = {};
  v8f acc[4];
#pragma unroll
  for (int s = 0; s < 4; ++s) acc[s] = zz;

  const int lr = tid >> 2, lc = (tid & 3) << 3;
  const int bc = tid >> 2, bk = (tid & 3) << 3;
  int hm = m0 + lr; if (hm >= cnt) hm = cnt - 1;
  const __bf16* aptr  = hbuf + (size_t)(hbase + hm) * FDIM + lc;
  const __bf16* dcol0 = down_bt + ((size_t)e * DDIM + n0 + bc) * FDIM + bk;
  const __bf16* dcol1 = dcol0 + (size_t)64 * FDIM;

  auto issue = [&](int k0, int p) {
    async_copy_b128(aptr + k0,  &As[p][lr][lc]);
    async_copy_b128(dcol0 + k0, &Bd[p][bc][bk]);
    async_copy_b128(dcol1 + k0, &Bd[p][bc + 64][bk]);
  };

  auto compute = [&](int p) {
    v16bf af = load_frag(&As[p][arow][kb]);
    v16bf bd[4];
#pragma unroll
    for (int s = 0; s < 4; ++s) {
      int bcc = (nw << 6) + (s << 4) + (lane & 15);
      bd[s] = load_frag(&Bd[p][bcc][kb]);
    }
#pragma unroll
    for (int s = 0; s < 4; ++s) {
      acc[s] = __builtin_amdgcn_wmma_f32_16x16x32_bf16(false, af, false, bd[s], (short)0, acc[s], false, false);
    }
  };

  constexpr int NK = FDIM / TK;   // 176, even
  issue(0, 0);
  for (int t = 0; t < NK - 2; t += 2) {
    issue((t + 1) * TK, 1);
    wait_async_le3();
    __syncthreads();
    compute(0);
    __syncthreads();
    issue((t + 2) * TK, 0);
    wait_async_le3();
    __syncthreads();
    compute(1);
    __syncthreads();
  }
  issue((NK - 1) * TK, 1);
  wait_async_le3();
  __syncthreads();
  compute(0);
  __syncthreads();
  wait_async_0();
  __syncthreads();
  compute(1);

#pragma unroll
  for (int s = 0; s < 4; ++s) {
    int col = n0 + (nw << 6) + (s << 4) + (lane & 15);
#pragma unroll
    for (int j = 0; j < 8; ++j) {
      int row = (mw << 4) + ((lane & 16) ? (j + 8) : j);
      int m = m0 + row;
      if (m < cnt) {
        // <=2 experts touch a token; fp32 add is commutative -> deterministic
        atomicAdd(out + (size_t)rowTok[row] * DDIM + col, acc[s][j] * rowWk[row]);
      }
    }
  }
}

extern "C" void kernel_launch(void* const* d_in, const int* in_sizes, int n_in,
                              void* d_out, int out_size, void* d_ws, size_t ws_size,
                              hipStream_t stream) {
  const float* x      = (const float*)d_in[0];   // [N, D]
  const float* rw     = (const float*)d_in[1];   // [N, 2]
  const int*   ridx   = (const int*)d_in[2];     // [N, 2]
  const float* gate_w = (const float*)d_in[3];   // [E, D, F]
  const float* up_w   = (const float*)d_in[4];   // [E, D, F]
  const float* down_w = (const float*)d_in[5];   // [E, F, D]
  float* out = (float*)d_out;

  char* w = (char*)d_ws;
  size_t o = 0;
  __bf16* xb      = (__bf16*)(w + o); o += (size_t)NTOK * DDIM * 2;          // 33.5 MB
  int* counts     = (int*)(w + o);    o += 256;
  int* offsets    = (int*)(w + o);    o += 256;
  int* tok_ids    = (int*)(w + o);    o += (size_t)NEXP * NTOK * 4;          // 256 KB
  float* wkbuf    = (float*)(w + o);  o += (size_t)NEXP * NTOK * 4;          // 256 KB
  __bf16* gate_bt = (__bf16*)(w + o); o += (size_t)NEXP * DDIM * FDIM * 2;   // 184 MB
  __bf16* up_bt   = (__bf16*)(w + o); o += (size_t)NEXP * DDIM * FDIM * 2;   // 184 MB
  __bf16* down_bt = (__bf16*)(w + o); o += (size_t)NEXP * DDIM * FDIM * 2;   // 184 MB
  __bf16* hbuf    = (__bf16*)(w + o);                                        // 184 MB

  // one-time transcode: bf16 + transpose to K-contiguous weight layouts
  dim3 tg1(FDIM / 32, DDIM / 32, NEXP);   // [D,F] -> [F,D]
  transcode_T_k<<<tg1, 256, 0, stream>>>(gate_w, gate_bt, DDIM, FDIM);
  transcode_T_k<<<tg1, 256, 0, stream>>>(up_w, up_bt, DDIM, FDIM);
  dim3 tg2(DDIM / 32, FDIM / 32, NEXP);   // [F,D] -> [D,F]
  transcode_T_k<<<tg2, 256, 0, stream>>>(down_w, down_bt, FDIM, DDIM);

  const size_t nxd4 = (size_t)NTOK * DDIM / 4;   // 4,194,304
  cast_x_k<<<(unsigned)(nxd4 / 256), 256, 0, stream>>>(x, xb, nxd4);
  zero_f4_k<<<(unsigned)(nxd4 / 256), 256, 0, stream>>>(out, nxd4);
  zero_i_k<<<1, 64, 0, stream>>>(counts, NEXP);
  routing_k<<<NTOK / 256, 256, 0, stream>>>(rw, ridx, counts, tok_ids, wkbuf);
  scan_k<<<1, 32, 0, stream>>>(counts, offsets);

  dim3 g1(FDIM / TN, NTOK / TM, NEXP);   // 44 x 128 x 8, early-exit past count
  gemm1_kernel<<<g1, 256, 0, stream>>>(xb, gate_bt, up_bt, counts, offsets, tok_ids, hbuf);

  dim3 g2(DDIM / TN, NTOK / TM, NEXP);   // 16 x 128 x 8
  gemm2_kernel<<<g2, 256, 0, stream>>>(hbuf, down_bt, counts, offsets, tok_ids, wkbuf, out);
}
